// StabilityPredictorSchnet_43009802502319
// MI455X (gfx1250) — compile-verified, matrix operand order checked by reading, not run on hardware
//
#include <hip/hip_runtime.h>
#include <hip/hip_bf16.h>
#include <math.h>

// Problem constants (from reference): B=4, L=4096, K=48, D=128
#define DB 4
#define DL 4096
#define DK 48
#define DD 128

typedef __attribute__((ext_vector_type(16))) _Float16 v16h;
typedef __attribute__((ext_vector_type(8)))  _Float16 v8h;
typedef __attribute__((ext_vector_type(8)))  float    v8f;
typedef __attribute__((ext_vector_type(4)))  float    f32x4;
typedef __attribute__((ext_vector_type(4)))  unsigned int u32x4;
typedef __attribute__((ext_vector_type(8)))  int      i32x8;
typedef __attribute__((ext_vector_type(4)))  int      i32x4;

// Address-space-qualified pointee types for the async-LDS builtin
typedef __attribute__((address_space(1))) i32x4 as1_i32x4;
typedef __attribute__((address_space(3))) i32x4 as3_i32x4;

#define WS 136   // LDS stride (halves) per weight column: 272B -> +4 banks/row, conflict-free b128 reads
#define AS 136   // LDS stride (halves) per A-tile row
#define XS 132   // LDS stride (floats) per gathered x_j row
#define WAVES 4  // waves per block; one (b,i) per wave

#if defined(__gfx1250__) && __has_builtin(__builtin_amdgcn_tensor_load_to_lds)
#define USE_TDM 1
#endif
#if defined(__gfx1250__) && __has_builtin(__builtin_amdgcn_global_load_async_to_lds_b128)
#define USE_ASYNC_GATHER 1
#endif

#define AS3P(p) ((__attribute__((address_space(3))) void*)(p))

// Branchless erf (Abramowitz-Stegun 7.1.26, |err| < 1.5e-7): rcp + exp2 + 5 FMA, no EXEC churn.
__device__ __forceinline__ float erf_fast(float x) {
    const float ax = fabsf(x);
    const float t  = __builtin_amdgcn_rcpf(fmaf(0.3275911f, ax, 1.0f));
    float p = fmaf(t, 1.061405429f, -1.453152027f);
    p = fmaf(t, p, 1.421413741f);
    p = fmaf(t, p, -0.284496736f);
    p = fmaf(t, p, 0.254829592f);
    p = p * t;
    const float e = __builtin_amdgcn_exp2f(-ax * ax * 1.4426950408889634f); // exp(-x^2)
    return copysignf(fmaf(-p, e, 1.0f), x);
}

__device__ __forceinline__ float gelu_exact(float x) {
    // 0.5*x*(1+erf(x/sqrt(2)))
    return 0.5f * x * (1.0f + erf_fast(x * 0.7071067811865475f));
}

// Build a 16-lane-striped 16-bit fragment (A 16x32 or B 32x16 layout) from LDS.
// Halves 0..7 <-> K = k0..k0+7, halves 8..15 <-> K = k0+16..k0+23, k0 = 32*kc + 8*(lane>=16).
__device__ __forceinline__ v16h ld_frag_lds(const _Float16* base, int stride, int rc, int k0) {
    union { v16h v; v8h h[2]; } u;
    u.h[0] = *(const v8h*)(base + rc * stride + k0);
    u.h[1] = *(const v8h*)(base + rc * stride + k0 + 16);
    return u.v;
}

#ifdef USE_TDM
// TDM: DMA a dense 2D fp32 tile [dim1 x dim0] (row stride = stride0 elements) into LDS.
// Descriptor per CDNA5 ISA 8.3/8.4: group0 {count=1, lds_addr, global_addr, type=2},
// group1 {data_size=4B, tensor_dim0/1, tile_dim0/1, tensor_dim0_stride}. Groups 2/3 zero (2D).
// amdgpu-toolchain (clang-23) 6-arg form: (u32x4, i32x8, i32x4, i32x4, i32x8, i32 cpol).
__device__ __forceinline__ void tdm_load_2d_f32(const float* gsrc, float* ldsDst,
                                                int dim0, int dim1, int stride0) {
    const unsigned long long ga = (unsigned long long)(uintptr_t)gsrc;
    const unsigned int laddr = (unsigned int)(uintptr_t)AS3P(ldsDst);
    u32x4 g0;
    g0[0] = 1u;                                                  // count=1 (valid user D#)
    g0[1] = laddr;                                               // lds_addr (bytes)
    g0[2] = (unsigned int)(ga & 0xffffffffu);                    // global_addr[31:0]
    g0[3] = (unsigned int)((ga >> 32) & 0x1ffffffu) | (2u << 30);// global_addr[56:32] | type=2
    i32x8 g1;
    g1[0] = (int)(2u << 16);                                     // data_size = 2 (4 bytes)
    g1[1] = (int)(((unsigned)dim0 & 0xffffu) << 16);             // tensor_dim0[15:0]
    g1[2] = (int)((((unsigned)dim0 >> 16) & 0xffffu) |
                  (((unsigned)dim1 & 0xffffu) << 16));           // dim0[31:16] | dim1[15:0]
    g1[3] = (int)((((unsigned)dim1 >> 16) & 0xffffu) |
                  (((unsigned)dim0 & 0xffffu) << 16));           // dim1[31:16] | tile_dim0
    g1[4] = (int)((unsigned)dim1 & 0xffffu);                     // tile_dim1 (tile_dim2 = 0)
    g1[5] = stride0;                                             // tensor_dim0_stride[31:0]
    g1[6] = 0;
    g1[7] = 0;
    const i32x4 z4 = {0, 0, 0, 0};
    const i32x8 z8 = {0, 0, 0, 0, 0, 0, 0, 0};
    __builtin_amdgcn_tensor_load_to_lds(g0, g1, z4, z4, z8, 0);
}
#endif

__device__ __forceinline__ void wait_async_lds() {
#ifdef USE_ASYNC_GATHER
#if __has_builtin(__builtin_amdgcn_s_wait_asynccnt)
    __builtin_amdgcn_s_wait_asynccnt(0);
#else
    asm volatile("s_wait_asynccnt 0" ::: "memory");
#endif
#endif
}

__global__ __launch_bounds__(128, 1)
void k_filter_conv(const float* __restrict__ hE, const float* __restrict__ hV,
                   const int* __restrict__ Eidx,
                   const float* __restrict__ fW1, const float* __restrict__ fb1,
                   const float* __restrict__ fW2, const float* __restrict__ fb2,
                   float* __restrict__ xconv) {
    __shared__ _Float16 sW1[DD * WS];          // fW1 transposed [n][k], f16
    __shared__ _Float16 sW2[DD * WS];          // fW2 transposed [n][k], f16
    __shared__ float    sB1[DD];
    __shared__ float    sB2[DD];
    __shared__ _Float16 sA[WAVES][16 * AS];    // per-wave GEMM1->GEMM2 staging (f16)
    __shared__ float    sX[WAVES][16 * XS];    // per-wave gathered h_V rows (f32)
#ifdef USE_TDM
    __shared__ float    sStage[DD * DD];       // TDM staging for raw fp32 weights (64 KB)
#endif

    const int tid  = threadIdx.x;
    const int wave = tid >> 5;
    const int lane = tid & 31;
    const int half = lane >> 4;
    const int l16  = lane & 15;

#ifdef USE_TDM
    // --- Tensor Data Mover path: DMA fp32 weights to LDS, then convert/transpose to f16 ---
    if (wave == 0) {
        tdm_load_2d_f32(fW1, sStage, DD, DD, DD);
        __builtin_amdgcn_s_wait_tensorcnt(0);
    }
    __syncthreads();
    for (int e = tid; e < DD * DD; e += 128) {
        const int k = e >> 7, n = e & 127;
        sW1[n * WS + k] = (_Float16)sStage[e];
    }
    __syncthreads();
    if (wave == 0) {
        tdm_load_2d_f32(fW2, sStage, DD, DD, DD);
        __builtin_amdgcn_s_wait_tensorcnt(0);
    }
    __syncthreads();
    for (int e = tid; e < DD * DD; e += 128) {
        const int k = e >> 7, n = e & 127;
        sW2[n * WS + k] = (_Float16)sStage[e];
    }
#else
    for (int e = tid; e < DD * DD; e += 128) {
        const int k = e >> 7, n = e & 127;     // fW[k][n] row-major
        sW1[n * WS + k] = (_Float16)fW1[e];
        sW2[n * WS + k] = (_Float16)fW2[e];
    }
#endif
    if (tid < DD) { sB1[tid] = fb1[tid]; sB2[tid] = fb2[tid]; }
    __syncthreads();

    const int bi = blockIdx.x * WAVES + wave;  // flattened (b,i), 0..B*L-1
    const int b  = bi >> 12;                   // L = 4096

    float csum[8];
    #pragma unroll
    for (int t = 0; t < 8; ++t) csum[t] = 0.f;

    for (int kt = 0; kt < 3; ++kt) {           // K = 48 = 3 tiles of 16 k-rows
        const long rowBase = (long)bi * DK + kt * 16;
        const float* aSrc = hE + (rowBase + l16) * (long)DD;
        if (kt < 2) __builtin_prefetch(aSrc + 16 * DD, 0, 1);   // next tile -> global_prefetch

        // ---- A1 fragments straight from global h_E (streaming, non-temporal) ----
        v16h a1[4];
        #pragma unroll
        for (int kc = 0; kc < 4; ++kc) {
            const float* p = aSrc + kc * 32 + 8 * half;
            union { f32x4 q[4]; float f[16]; } tf;
            tf.q[0] = __builtin_nontemporal_load((const f32x4*)(p));
            tf.q[1] = __builtin_nontemporal_load((const f32x4*)(p + 4));
            tf.q[2] = __builtin_nontemporal_load((const f32x4*)(p + 16));
            tf.q[3] = __builtin_nontemporal_load((const f32x4*)(p + 20));
            union { v16h v; _Float16 e[16]; } u;
            #pragma unroll
            for (int j = 0; j < 16; ++j) u.e[j] = (_Float16)tf.f[j];
            a1[kc] = u.v;
        }

        // ---- GEMM1: T1 = gelu(hE @ fW1 + fb1); stage f16 tile to LDS ----
        #pragma unroll
        for (int t = 0; t < 8; ++t) {
            v8f acc = {};
            #pragma unroll
            for (int kc = 0; kc < 4; ++kc) {
                v16h bf = ld_frag_lds(sW1, WS, 16 * t + l16, kc * 32 + 8 * half);
                acc = __builtin_amdgcn_wmma_f32_16x16x32_f16(false, a1[kc], false, bf,
                                                             (short)0, acc, false, false);
            }
            const float bias = sB1[16 * t + l16];
            #pragma unroll
            for (int r = 0; r < 8; ++r) {
                const float v = gelu_exact(acc[r] + bias);
                sA[wave][(r + 8 * half) * AS + 16 * t + l16] = (_Float16)v;  // C->A relayout
            }
        }

        // ---- gather 16 neighbor rows x_j = h_V[b, E_idx[...]] into LDS ----
        #pragma unroll 4
        for (int r = 0; r < 16; ++r) {
            const int nb = Eidx[rowBase + r];
            const float* gsrc = hV + ((long)b * DL + nb) * (long)DD + 4 * lane;
#ifdef USE_ASYNC_GATHER
            // one instruction per row: 16B/lane x 32 lanes = full 512B row, on ASYNCcnt
            __builtin_amdgcn_global_load_async_to_lds_b128(
                (as1_i32x4*)gsrc, (as3_i32x4*)&sX[wave][r * XS + 4 * lane], 0, 0);
#else
            *(f32x4*)&sX[wave][r * XS + 4 * lane] = *(const f32x4*)gsrc;
#endif
        }
        __syncthreads();

        // ---- A2 fragments from staged T1 ----
        v16h a2[4];
        #pragma unroll
        for (int kc = 0; kc < 4; ++kc)
            a2[kc] = ld_frag_lds(sA[wave], AS, l16, kc * 32 + 8 * half);

        wait_async_lds();   // gathered rows overlap A2 fragment loads / GEMM2 issue

        // ---- GEMM2 + gelu -> W, fused (W * x_j) and reduction over the 16 k-rows ----
        #pragma unroll
        for (int t = 0; t < 8; ++t) {
            v8f acc = {};
            #pragma unroll
            for (int kc = 0; kc < 4; ++kc) {
                v16h bf = ld_frag_lds(sW2, WS, 16 * t + l16, kc * 32 + 8 * half);
                acc = __builtin_amdgcn_wmma_f32_16x16x32_f16(false, a2[kc], false, bf,
                                                             (short)0, acc, false, false);
            }
            const float bias = sB2[16 * t + l16];
            float s = 0.f;
            #pragma unroll
            for (int r = 0; r < 8; ++r) {
                const float w = gelu_exact(acc[r] + bias);
                s = fmaf(w, sX[wave][(r + 8 * half) * XS + 16 * t + l16], s);
            }
            csum[t] += s;
        }
        __syncthreads();
    }

    // merge the two half-wave row groups (wave32): lanes l and l+16 hold complementary rows
    #pragma unroll
    for (int t = 0; t < 8; ++t) csum[t] += __shfl_xor(csum[t], 16);
    if (lane < 16) {
        #pragma unroll
        for (int t = 0; t < 8; ++t)
            xconv[(long)bi * DD + 16 * t + lane] = csum[t];
    }
}

__global__ __launch_bounds__(128, 1)
void k_mlp_reduce(const float* __restrict__ xconv,
                  const float* __restrict__ mW1, const float* __restrict__ mb1,
                  const float* __restrict__ mW2, const float* __restrict__ mb2,
                  const float* __restrict__ mW3, const float* __restrict__ mb3,
                  const float* __restrict__ mask, float* __restrict__ acc) {
    __shared__ float sx[DD];
    __shared__ float sy[DD];
    const int tid = threadIdx.x;
    const int row = blockIdx.x;        // 0..B*L-1
    const int b   = row >> 12;

    sx[tid] = xconv[(long)row * DD + tid];
    __syncthreads();

    float s = 0.f;
    #pragma unroll 8
    for (int k = 0; k < DD; ++k) s = fmaf(sx[k], mW1[k * DD + tid], s);
    sy[tid] = gelu_exact(s + mb1[tid]);
    __syncthreads();

    float partial = 0.f;
    if (tid < 64) {
        float s2 = 0.f;
        #pragma unroll 8
        for (int k = 0; k < DD; ++k) s2 = fmaf(sy[k], mW2[k * 64 + tid], s2);
        partial = gelu_exact(s2 + mb2[tid]) * mW3[tid];
    }
    __syncthreads();
    sx[tid] = partial;
    __syncthreads();

    if (tid == 0) {
        float dg = mb3[0];
        for (int k = 0; k < 64; ++k) dg += sx[k];
        const float m = mask[row];
        atomicAdd(&acc[b], dg * m);
        atomicAdd(&acc[DB + b], m);
    }
}

__global__ void k_zero_acc(float* __restrict__ acc) {
    if (threadIdx.x < 2 * DB) acc[threadIdx.x] = 0.f;
}

__global__ void k_finish(const float* __restrict__ acc, float* __restrict__ out) {
    const int t = threadIdx.x;
    if (t < DB) out[t] = acc[t] / sqrtf(fmaxf(acc[DB + t], 1.0f));
}

extern "C" void kernel_launch(void* const* d_in, const int* in_sizes, int n_in,
                              void* d_out, int out_size, void* d_ws, size_t ws_size,
                              hipStream_t stream) {
    (void)in_sizes; (void)n_in; (void)out_size; (void)ws_size;
    const float* hV   = (const float*)d_in[0];
    const float* hE   = (const float*)d_in[1];
    const int*   Eidx = (const int*)d_in[2];
    const float* mask = (const float*)d_in[3];
    const float* fW1  = (const float*)d_in[4];
    const float* fb1  = (const float*)d_in[5];
    const float* fW2  = (const float*)d_in[6];
    const float* fb2  = (const float*)d_in[7];
    const float* mW1  = (const float*)d_in[8];
    const float* mb1  = (const float*)d_in[9];
    const float* mW2  = (const float*)d_in[10];
    const float* mb2  = (const float*)d_in[11];
    const float* mW3  = (const float*)d_in[12];
    const float* mb3  = (const float*)d_in[13];
    float* out = (float*)d_out;

    float* xconv = (float*)d_ws;                       // [B*L, D] f32 (8 MB)
    float* acc   = xconv + (size_t)DB * DL * DD;       // [2*B] f32 accumulators

    k_zero_acc<<<1, 32, 0, stream>>>(acc);
    k_filter_conv<<<(DB * DL) / WAVES, 32 * WAVES, 0, stream>>>(hE, hV, Eidx,
                                                                fW1, fb1, fW2, fb2, xconv);
    k_mlp_reduce<<<DB * DL, 128, 0, stream>>>(xconv, mW1, mb1, mW2, mb2, mW3, mb3, mask, acc);
    k_finish<<<1, 32, 0, stream>>>(acc, out);
}
